// SAGE_89996744720665
// MI455X (gfx1250) — matrix operand order repaired and checked by reading, not compiled, not run on hardware
//
#include <hip/hip_runtime.h>

typedef __attribute__((ext_vector_type(16))) _Float16 v16h;
typedef __attribute__((ext_vector_type(8)))  float    v8f;

// ---------------------------------------------------------------------------
// Zero-fill (grid-stride)
// ---------------------------------------------------------------------------
__global__ __launch_bounds__(256) void sage_zero_f32(float* __restrict__ p, long long n) {
  long long i = (long long)blockIdx.x * blockDim.x + threadIdx.x;
  long long stride = (long long)gridDim.x * blockDim.x;
  for (; i < n; i += stride) p[i] = 0.0f;
}

// ---------------------------------------------------------------------------
// Weight prep: W [Kd, 256] f32  ->  Wt [256, Kd] f16 (transpose + convert).
// Makes each WMMA B-fragment a single contiguous 32B vector load.
// ---------------------------------------------------------------------------
__global__ __launch_bounds__(256) void sage_wt_prep(const float* __restrict__ W,
                                                    _Float16* __restrict__ Wt,
                                                    int Kd) {
  int idx = blockIdx.x * blockDim.x + threadIdx.x;
  int total = Kd * 256;
  if (idx >= total) return;
  int k = idx >> 8;          // idx / 256  (coalesced read of W)
  int n = idx & 255;         // idx % 256
  Wt[n * Kd + k] = (_Float16)W[idx];
}

// ---------------------------------------------------------------------------
// Degree: deg[dst[e]] += 1  (global_atomic_add_f32)
// ---------------------------------------------------------------------------
__global__ __launch_bounds__(256) void sage_degree(const int* __restrict__ dst,
                                                   float* __restrict__ deg, int E) {
  int e = blockIdx.x * blockDim.x + threadIdx.x;
  if (e < E) atomicAdd(&deg[dst[e]], 1.0f);
}

// ---------------------------------------------------------------------------
// Edge-parallel scatter-add: agg[dst[e], :] += Hin[src[e], :]
// float4 feature chunk is the fast dimension across consecutive threads,
// so gathers and atomics are contiguous per edge (L2-resident on MI455X).
// ---------------------------------------------------------------------------
__global__ __launch_bounds__(256) void sage_scatter(const float* __restrict__ Hin,
                                                    const int* __restrict__ src,
                                                    const int* __restrict__ dst,
                                                    float* __restrict__ agg,
                                                    int E, int d) {
  const int chunks = d >> 2;
  long long idx = (long long)blockIdx.x * blockDim.x + threadIdx.x;
  long long total = (long long)E * chunks;
  if (idx >= total) return;
  int e = (int)(idx / chunks);
  int f = ((int)(idx % chunks)) << 2;
  int s = src[e];
  int t = dst[e];
  const float4 v = *reinterpret_cast<const float4*>(Hin + (long long)s * d + f);
  float* o = agg + (long long)t * d + f;
  atomicAdd(o + 0, v.x);
  atomicAdd(o + 1, v.y);
  atomicAdd(o + 2, v.z);
  atomicAdd(o + 3, v.w);
}

// ---------------------------------------------------------------------------
// One k-pass: A from f32 rows (scaled, converted on the fly), B from
// pre-converted f16 transposed weights (contiguous 32B per fragment).
// A fragment is reused by 4 WMMAs covering 64 output columns.
// ---------------------------------------------------------------------------
template <int KD>
__device__ __forceinline__ void sage_k_pass(
    const float* __restrict__ Msrc,      // [N, KD] source rows (f32)
    const float scale,                   // per-row scale folded into A cvt
    const _Float16* __restrict__ Wt,     // [256, KD] f16, transposed weights
    const int row_g, const int a_kb, const int b_kb, const int colbase,
    v8f& c0, v8f& c1, v8f& c2, v8f& c3)
{
  const float*    arow  = Msrc + (long long)row_g * KD + a_kb;
  const _Float16* wcol0 = Wt + (long long)(colbase +  0) * KD + b_kb;
  const _Float16* wcol1 = Wt + (long long)(colbase + 16) * KD + b_kb;
  const _Float16* wcol2 = Wt + (long long)(colbase + 32) * KD + b_kb;
  const _Float16* wcol3 = Wt + (long long)(colbase + 48) * KD + b_kb;

#pragma unroll
  for (int kt = 0; kt < KD; kt += 32) {
    // ---- A fragment (16-bit A 16x32 layout): 4 x float4 + cvt ----
    const float4 f0 = *reinterpret_cast<const float4*>(arow + kt);
    const float4 f1 = *reinterpret_cast<const float4*>(arow + kt + 4);
    const float4 f2 = *reinterpret_cast<const float4*>(arow + kt + 8);
    const float4 f3 = *reinterpret_cast<const float4*>(arow + kt + 12);
    v16h a;
    a[0]  = (_Float16)(f0.x * scale); a[1]  = (_Float16)(f0.y * scale);
    a[2]  = (_Float16)(f0.z * scale); a[3]  = (_Float16)(f0.w * scale);
    a[4]  = (_Float16)(f1.x * scale); a[5]  = (_Float16)(f1.y * scale);
    a[6]  = (_Float16)(f1.z * scale); a[7]  = (_Float16)(f1.w * scale);
    a[8]  = (_Float16)(f2.x * scale); a[9]  = (_Float16)(f2.y * scale);
    a[10] = (_Float16)(f2.z * scale); a[11] = (_Float16)(f2.w * scale);
    a[12] = (_Float16)(f3.x * scale); a[13] = (_Float16)(f3.y * scale);
    a[14] = (_Float16)(f3.z * scale); a[15] = (_Float16)(f3.w * scale);

    // ---- 4 B fragments: one 32B contiguous f16 load each, no cvt ----
    const v16h b0 = *reinterpret_cast<const v16h*>(wcol0 + kt);
    const v16h b1 = *reinterpret_cast<const v16h*>(wcol1 + kt);
    const v16h b2 = *reinterpret_cast<const v16h*>(wcol2 + kt);
    const v16h b3 = *reinterpret_cast<const v16h*>(wcol3 + kt);

    c0 = __builtin_amdgcn_wmma_f32_16x16x32_f16(false, a, false, b0, (short)0, c0, false, false);
    c1 = __builtin_amdgcn_wmma_f32_16x16x32_f16(false, a, false, b1, (short)0, c1, false, false);
    c2 = __builtin_amdgcn_wmma_f32_16x16x32_f16(false, a, false, b2, (short)0, c2, false, false);
    c3 = __builtin_amdgcn_wmma_f32_16x16x32_f16(false, a, false, b3, (short)0, c3, false, false);
  }
}

// ---------------------------------------------------------------------------
// Fused SAGE GEMM with WMMA: each wave computes a 16x64 output tile.
//   Hout = act( (agg/deg) @ Wl + bl + Hin @ Wr )
// ---------------------------------------------------------------------------
template <int KD, bool RELU>
__global__ __launch_bounds__(256) void sage_gemm_wmma(
    const float* __restrict__ Hin,        // [N, KD]
    const float* __restrict__ agg,        // [N, KD]
    const float* __restrict__ deg,        // [N]
    const _Float16* __restrict__ Wlt,     // [256, KD] f16 transposed
    const float* __restrict__ bl,         // [256]
    const _Float16* __restrict__ Wrt,     // [256, KD] f16 transposed
    float* __restrict__ Hout,             // [N, 256]
    int Nnodes)
{
  const int NOUT = 256;

  const int wave = (int)((blockIdx.x * blockDim.x + threadIdx.x) >> 5);
  const int lane = threadIdx.x & 31;
  const int tile_m = wave >> 2;         // 4 column groups per m-tile
  const int grp_n  = wave & 3;
  if (tile_m * 16 >= Nnodes) return;

  const int half_lo = (lane < 16);
  const int row_g   = tile_m * 16 + (lane & 15);
  const int a_kb    = half_lo ? 0 : 16;   // A lane K-base (contiguous k = 16h+s, matches B)
  const int b_kb    = half_lo ? 0 : 16;   // B lane K-base (32x16 f16 B)
  const int colbase = grp_n * 64 + (lane & 15);

  const float invdeg = 1.0f / fmaxf(deg[row_g], 1.0f);

  v8f c0 = {}, c1 = {}, c2 = {}, c3 = {};

  // mean part: (agg * invdeg) @ Wl
  sage_k_pass<KD>(agg, invdeg, Wlt, row_g, a_kb, b_kb, colbase, c0, c1, c2, c3);
  // root part: Hin @ Wr
  sage_k_pass<KD>(Hin, 1.0f,  Wrt, row_g, a_kb, b_kb, colbase, c0, c1, c2, c3);

  // ---- epilogue: bias (+ ReLU), straight-line stores (N % 16 == 0) ----
  const int m0 = tile_m * 16 + (half_lo ? 0 : 8);
  v8f* accs[4] = {&c0, &c1, &c2, &c3};
#pragma unroll
  for (int nt = 0; nt < 4; ++nt) {
    const int col = colbase + nt * 16;
    const float bias = bl[col];
    v8f& c = *accs[nt];
#pragma unroll
    for (int r = 0; r < 8; ++r) {
      float v = c[r] + bias;
      if (RELU) v = fmaxf(v, 0.0f);
      Hout[(long long)(m0 + r) * NOUT + col] = v;
    }
  }
}

// ---------------------------------------------------------------------------
// Launch: 3 layers. inputs: x, edge_index, Wl0,bl0,Wr0, Wl1,bl1,Wr1, Wl2,bl2,Wr2
// ---------------------------------------------------------------------------
extern "C" void kernel_launch(void* const* d_in, const int* in_sizes, int n_in,
                              void* d_out, int out_size, void* d_ws, size_t ws_size,
                              hipStream_t stream) {
  const float* x   = (const float*)d_in[0];
  const int*   ei  = (const int*)d_in[1];
  const float* Wl0 = (const float*)d_in[2];
  const float* bl0 = (const float*)d_in[3];
  const float* Wr0 = (const float*)d_in[4];
  const float* Wl1 = (const float*)d_in[5];
  const float* bl1 = (const float*)d_in[6];
  const float* Wr1 = (const float*)d_in[7];
  const float* Wl2 = (const float*)d_in[8];
  const float* bl2 = (const float*)d_in[9];
  const float* Wr2 = (const float*)d_in[10];

  const int D_IN = 128, D_H = 256;
  const int E = in_sizes[1] / 2;
  const int N = in_sizes[0] / D_IN;
  const int* src = ei;
  const int* dst = ei + E;

  float* out = (float*)d_out;

  // ---- workspace carve-up ----
  float* ws  = (float*)d_ws;
  float* deg = ws;                               // [N]
  float* agg = deg + N;                          // [N, 256] (max d)
  float* h1  = agg + (long long)N * D_H;         // [N, 256]
  float* h2  = h1  + (long long)N * D_H;         // [N, 256]
  _Float16* whalf = (_Float16*)(h2 + (long long)N * D_H);
  _Float16* Wl0t = whalf;                        // [256, 128]
  _Float16* Wr0t = Wl0t + 256 * D_IN;            // [256, 128]
  _Float16* Wl1t = Wr0t + 256 * D_IN;            // [256, 256]
  _Float16* Wr1t = Wl1t + 256 * D_H;             // [256, 256]
  _Float16* Wl2t = Wr1t + 256 * D_H;             // [256, 256]
  _Float16* Wr2t = Wl2t + 256 * D_H;             // [256, 256]

  const int TB = 256;

  // ---- weight prep (transpose + f32->f16) ----
  sage_wt_prep<<<(D_IN * 256 + TB - 1) / TB, TB, 0, stream>>>(Wl0, Wl0t, D_IN);
  sage_wt_prep<<<(D_IN * 256 + TB - 1) / TB, TB, 0, stream>>>(Wr0, Wr0t, D_IN);
  sage_wt_prep<<<(D_H  * 256 + TB - 1) / TB, TB, 0, stream>>>(Wl1, Wl1t, D_H);
  sage_wt_prep<<<(D_H  * 256 + TB - 1) / TB, TB, 0, stream>>>(Wr1, Wr1t, D_H);
  sage_wt_prep<<<(D_H  * 256 + TB - 1) / TB, TB, 0, stream>>>(Wl2, Wl2t, D_H);
  sage_wt_prep<<<(D_H  * 256 + TB - 1) / TB, TB, 0, stream>>>(Wr2, Wr2t, D_H);

  // ---- degree once (dst is fixed across layers) ----
  sage_zero_f32<<<512, TB, 0, stream>>>(deg, (long long)N);
  sage_degree<<<(E + TB - 1) / TB, TB, 0, stream>>>(dst, deg, E);

  const int tiles_m = (N + 15) / 16;
  const int gblocks = (tiles_m * 4 + 7) / 8;   // 4 col-groups/tile, 8 waves/block

  // ---- layer 0: K=128, relu ----
  sage_zero_f32<<<2048, TB, 0, stream>>>(agg, (long long)N * D_IN);
  {
    long long total = (long long)E * (D_IN / 4);
    sage_scatter<<<(int)((total + TB - 1) / TB), TB, 0, stream>>>(x, src, dst, agg, E, D_IN);
  }
  sage_gemm_wmma<128, true><<<gblocks, TB, 0, stream>>>(x, agg, deg, Wl0t, bl0, Wr0t, h1, N);

  // ---- layer 1: K=256, relu ----
  sage_zero_f32<<<2048, TB, 0, stream>>>(agg, (long long)N * D_H);
  {
    long long total = (long long)E * (D_H / 4);
    sage_scatter<<<(int)((total + TB - 1) / TB), TB, 0, stream>>>(h1, src, dst, agg, E, D_H);
  }
  sage_gemm_wmma<256, true><<<gblocks, TB, 0, stream>>>(h1, agg, deg, Wl1t, bl1, Wr1t, h2, N);

  // ---- layer 2: K=256, no relu ----
  sage_zero_f32<<<2048, TB, 0, stream>>>(agg, (long long)N * D_H);
  {
    long long total = (long long)E * (D_H / 4);
    sage_scatter<<<(int)((total + TB - 1) / TB), TB, 0, stream>>>(h2, src, dst, agg, E, D_H);
  }
  sage_gemm_wmma<256, false><<<gblocks, TB, 0, stream>>>(h2, agg, deg, Wl2t, bl2, Wr2t, out, N);
}
